// TriangleAttentionStartingNode_20753281974320
// MI455X (gfx1250) — matrix operand order repaired and checked by reading, not compiled, not run on hardware
//
#include <hip/hip_runtime.h>

// ---------------------------------------------------------------------------
// TriangleAttentionStartingNode for gfx1250 (MI455X), wave32 + WMMA + TDM.
// All GEMM contractions use v_wmma_f32_16x16x32_f16 (f16 A/B, f32 accum).
// attn_kernel stages K / V^T into LDS via TENSOR_LOAD_TO_LDS (TDM) when the
// builtin is available (TENSORcnt-tracked DMA, no VGPR round trip).
// Workspace layout (requires ~84 MB of d_ws):
//   [0, 256KB)        packed f16 weights (B-matrix VGPR layout)
//   +0MB  Qw   f16 [65536][128]   (pre-scaled by 1/sqrt(32))
//   +16MB Kw   f16 [65536][128]
//   +32MB VTw  f16 [i][h][d][k]   (transposed V per (i,h))
//   +48MB Gw   f16 [65536][128]   (sigmoid gate)
//   +64MB valsw f16 [65536][128]  (gated attention output)
//   +80MB biasw f32 [h][j][k]     (x @ Wb, 1MB)
// ---------------------------------------------------------------------------

typedef __attribute__((ext_vector_type(16))) _Float16 v16h;
typedef __attribute__((ext_vector_type(8)))  _Float16 v8h;
typedef __attribute__((ext_vector_type(8)))  float    v8f;
typedef __attribute__((ext_vector_type(4)))  float    v4f;
typedef unsigned int u32x4 __attribute__((ext_vector_type(4)));
typedef int          i32x4 __attribute__((ext_vector_type(4)));
typedef int          i32x8 __attribute__((ext_vector_type(8)));

#define CZ     128
#define NSEQ   256
#define NPOS   65536          // 256*256
#define QSCALE 0.17677669529663689f   // 1/sqrt(32)

// packed-weight bases (in halves): Q,K,V,G,O,B
#define PK_Q 0
#define PK_K 16384
#define PK_V 32768
#define PK_G 49152
#define PK_O 65536
#define PK_B 81920

#if __has_builtin(__builtin_amdgcn_tensor_load_to_lds) && \
    __has_builtin(__builtin_amdgcn_s_wait_tensorcnt)
#define USE_TDM 1
#else
#define USE_TDM 0
#endif

// -------------------- cross-lane helpers (wave32) --------------------------
template <int MASK>
__device__ __forceinline__ float swz(float x) {
  // ds_swizzle_b32, group-of-32 mode: and=0x1f, or=0, xor=MASK
  return __int_as_float(
      __builtin_amdgcn_ds_swizzle(__float_as_int(x), (MASK << 10) | 0x1f));
}
__device__ __forceinline__ float red16max(float x) {
  x = fmaxf(x, swz<1>(x)); x = fmaxf(x, swz<2>(x));
  x = fmaxf(x, swz<4>(x)); x = fmaxf(x, swz<8>(x));
  return x;
}
__device__ __forceinline__ float red16sum(float x) {
  x += swz<1>(x); x += swz<2>(x); x += swz<4>(x); x += swz<8>(x);
  return x;
}

// -------------------- WMMA fragment helpers --------------------------------
__device__ __forceinline__ v8f wmma16(v16h a, v16h b, v8f c) {
  return __builtin_amdgcn_wmma_f32_16x16x32_f16(false, a, false, b,
                                                (short)0, c, false, false);
}

// A fragment: row-major tile (row stride `stride` halves), K-chunk base k0.
// Lane layout (contiguous, matches B): lane<16: K = k0+[0..15]
//                                      lane>=16: K = k0+16+[0..15]
__device__ __forceinline__ v16h load_A16(const _Float16* rowbase, int stride,
                                         int k0, int lane) {
  int m = lane & 15, hi = lane >> 4;
  const _Float16* p = rowbase + m * stride + k0 + hi * 16;
  v8h lo = *(const v8h*)p;
  v8h hh = *(const v8h*)(p + 8);
  v16h a;
#pragma unroll
  for (int e = 0; e < 8; ++e) { a[e] = lo[e]; a[8 + e] = hh[e]; }
  return a;
}

// B fragment: 16 contiguous halves per lane (pre-arranged source).
__device__ __forceinline__ v16h load16(const _Float16* p) {
  v8h lo = *(const v8h*)p;
  v8h hh = *(const v8h*)(p + 8);
  v16h b;
#pragma unroll
  for (int e = 0; e < 8; ++e) { b[e] = lo[e]; b[8 + e] = hh[e]; }
  return b;
}

#if USE_TDM
// Issue a 2D TENSOR_LOAD_TO_LDS: tile0 elements per row (f16), tile1 rows,
// row stride `stride0` elements.  D# packing per CDNA5 ISA 8.3/8.4:
//   g0: count=1 | lds_addr | global_addr(57b) | type=2
//   g1: data_size=2B | tensor_dim0/1 = tile dims | tile_dim0/1 | dim0 stride
__device__ __forceinline__ void tdm_load_2d(unsigned lds_byte_addr,
                                            const void* gptr,
                                            unsigned tile0, unsigned tile1,
                                            unsigned long long stride0) {
  unsigned long long ga = (unsigned long long)(size_t)gptr;
  u32x4 g0;
  g0[0] = 1u;                                             // count=1
  g0[1] = lds_byte_addr;                                  // lds_addr
  g0[2] = (unsigned)ga;                                   // global_addr[31:0]
  g0[3] = ((unsigned)(ga >> 32) & 0x1FFFFFFu) | (2u << 30); // [56:32] | type=2
  i32x8 g1;
  g1[0] = 1 << 16;                                        // data_size = 2B
  g1[1] = (int)((tile0 & 0xFFFFu) << 16);                 // tensor_dim0 lo
  g1[2] = (int)(((tile0 >> 16) & 0xFFFFu) | ((tile1 & 0xFFFFu) << 16));
  g1[3] = (int)(((tile1 >> 16) & 0xFFFFu) | ((tile0 & 0xFFFFu) << 16)); // tile_dim0
  g1[4] = (int)(tile1 & 0xFFFFu);                         // tile_dim1, tile_dim2=0
  g1[5] = (int)(unsigned)(stride0 & 0xFFFFFFFFull);       // dim0 stride lo
  g1[6] = (int)((unsigned)(stride0 >> 32) & 0xFFFFu);     // dim0 stride hi
  g1[7] = 0;                                              // dim1 stride = 0
  i32x4 z4 = {0, 0, 0, 0};
#if __clang_major__ >= 23
  i32x8 z8 = {0, 0, 0, 0, 0, 0, 0, 0};
  __builtin_amdgcn_tensor_load_to_lds(g0, g1, z4, z4, z8, 0);
#else
  __builtin_amdgcn_tensor_load_to_lds(g0, g1, z4, z4, 0);
#endif
}
#endif

// -------------------- 1) weight packing ------------------------------------
// Packed element (nt,kt,lane,e) = W[k][n], n = nt*16 + (lane&15),
// k = kt*32 + e + 16*(lane>>4). One v16h per lane per (nt,kt) tile.
__global__ void pack_kernel(const float* __restrict__ Wq,
                            const float* __restrict__ Wk,
                            const float* __restrict__ Wv,
                            const float* __restrict__ Wg,
                            const float* __restrict__ Wo,
                            const float* __restrict__ Wb,
                            _Float16* __restrict__ packed) {
  int gid = blockIdx.x * 256 + threadIdx.x;   // 6 * 16384 slots
  int mi = gid >> 14;
  int r  = gid & 16383;
  if (mi > 5) return;
  if (mi == 5 && r >= 2048) return;           // Wb: single nt tile
  int e    = r & 15;
  int lane = (r >> 4) & 31;
  int kt   = (r >> 9) & 3;
  int nt   = r >> 11;
  int hi   = lane >> 4;
  int nn   = nt * 16 + (lane & 15);
  int k    = kt * 32 + e + 16 * hi;
  float val;
  int base;
  if (mi < 5) {
    const float* Ws[5] = {Wq, Wk, Wv, Wg, Wo};
    const int bases[5] = {PK_Q, PK_K, PK_V, PK_G, PK_O};
    val  = Ws[mi][k * 128 + nn];
    base = bases[mi];
  } else {
    val  = (nn < 4) ? Wb[k * 4 + nn] : 0.0f;
    base = PK_B;
  }
  packed[base + r] = (_Float16)val;
}

// -------------------- 2) LayerNorm + QKVG/bias projections -----------------
__global__ void __launch_bounds__(128)
lnproj_kernel(const float* __restrict__ x, const float* __restrict__ gamma,
              const float* __restrict__ beta,
              const _Float16* __restrict__ packed,
              const float* __restrict__ bg,
              _Float16* __restrict__ Qw, _Float16* __restrict__ Kw,
              _Float16* __restrict__ VTw, _Float16* __restrict__ Gw,
              float* __restrict__ biasw) {
  __shared__ _Float16 Xsh[4][16][CZ];         // 16 KB, per-wave tiles
  int wave = threadIdx.x >> 5;
  int lane = threadIdx.x & 31;
  int tile = blockIdx.x * 4 + wave;           // 4096 tiles of 16 rows
  int rowbase = tile * 16;
  int r = lane & 15, hi = lane >> 4, n = lane & 15;
  int row = rowbase + r;

  // ---- LayerNorm: 2 lanes per row, 64 f32 each ----
  const float* xp = x + (size_t)row * CZ + hi * 64;
  float s = 0.f, sq = 0.f;
  v4f xv[16];
#pragma unroll
  for (int t = 0; t < 16; ++t) {
    xv[t] = *(const v4f*)(xp + t * 4);
#pragma unroll
    for (int e = 0; e < 4; ++e) { float f = xv[t][e]; s += f; sq += f * f; }
  }
  s  += swz<16>(s);
  sq += swz<16>(sq);
  float mu   = s * (1.f / 128.f);
  float var  = sq * (1.f / 128.f) - mu * mu;
  float rstd = rsqrtf(var + 1e-5f);
#pragma unroll
  for (int t = 0; t < 16; ++t) {
#pragma unroll
    for (int e = 0; e < 4; ++e) {
      int c = hi * 64 + t * 4 + e;
      float xn = (xv[t][e] - mu) * rstd * gamma[c] + beta[c];
      Xsh[wave][r][c] = (_Float16)xn;
    }
  }
  __syncthreads();
  const _Float16* Xrow = &Xsh[wave][0][0];

  // ---- projections: Y = Xn @ W  (K=128 => 4 WMMA steps, N=128 => 8 tiles)
  const _Float16* bases[4] = {packed + PK_Q, packed + PK_K,
                              packed + PK_V, packed + PK_G};
#pragma unroll
  for (int m = 0; m < 4; ++m) {
    const _Float16* pw = bases[m];
#pragma unroll
    for (int nt = 0; nt < 8; ++nt) {
      v8f acc = {};
#pragma unroll
      for (int kt = 0; kt < 4; ++kt) {
        v16h a = load_A16(Xrow, CZ, kt * 32, lane);
        v16h b = load16(pw + ((size_t)(nt * 4 + kt) * 32 + lane) * 16);
        acc = wmma16(a, b, acc);
      }
      int col = nt * 16 + n;
#pragma unroll
      for (int v = 0; v < 8; ++v) {
        int prow = rowbase + v + 8 * hi;
        float val = acc[v];
        if (m == 0) {
          Qw[(size_t)prow * 128 + col] = (_Float16)(val * QSCALE);
        } else if (m == 1) {
          Kw[(size_t)prow * 128 + col] = (_Float16)val;
        } else if (m == 2) {
          int i = prow >> 8, kk = prow & 255;
          int h = col >> 5, d = col & 31;
          VTw[(((size_t)(i * 4 + h) * 32 + d) << 8) + kk] = (_Float16)val;
        } else {
          float gz = val + bg[col];
          float gs = 1.f / (1.f + __expf(-gz));
          Gw[(size_t)prow * 128 + col] = (_Float16)gs;
        }
      }
    }
  }

  // ---- attention bias head: x @ Wb (4 valid columns, zero-padded tile)
  {
    v8f acc = {};
#pragma unroll
    for (int kt = 0; kt < 4; ++kt) {
      v16h a = load_A16(Xrow, CZ, kt * 32, lane);
      v16h b = load16(packed + PK_B + ((size_t)(kt * 32 + lane)) * 16);
      acc = wmma16(a, b, acc);
    }
    if (n < 4) {
#pragma unroll
      for (int v = 0; v < 8; ++v) {
        int prow = rowbase + v + 8 * hi;       // prow = j*256 + k
        biasw[(size_t)n * NPOS + prow] = acc[v];
      }
    }
  }
}

// -------------------- 3) attention: softmax(QK^T/sqrt(d)+bias) @ V, gated --
__global__ void __launch_bounds__(64)
attn_kernel(const _Float16* __restrict__ Qw, const _Float16* __restrict__ Kw,
            const _Float16* __restrict__ VTw, const _Float16* __restrict__ Gw,
            const float* __restrict__ biasw, _Float16* __restrict__ valsw) {
  __shared__ _Float16 Ksh[NSEQ][32];      // [k][d]   16 KB
  __shared__ _Float16 VTsh[32][NSEQ];     // [d][k]   16 KB
  __shared__ _Float16 Psh[2][16][NSEQ];   // per-wave P tile, 16 KB
  int tid = threadIdx.x;                  // 64 threads = 2 waves
  int wave = tid >> 5, lane = tid & 31;
  int blk = blockIdx.x;                   // i*4 + h
  int i = blk >> 2, h = blk & 3;
  int n = lane & 15, hi = lane >> 4;

#if USE_TDM
  // ---- TDM: DMA K tile (32-elem rows, stride 128, 256 rows) and the
  //      contiguous 16KB V^T block straight into LDS; TENSORcnt-tracked.
  if (wave == 0) {
    tdm_load_2d((unsigned)(size_t)&Ksh[0][0],
                Kw + (size_t)(i * NSEQ) * 128 + h * 32,
                /*tile0=*/32, /*tile1=*/NSEQ, /*stride0=*/128ull);
    tdm_load_2d((unsigned)(size_t)&VTsh[0][0],
                VTw + (size_t)(i * 4 + h) * 32 * NSEQ,
                /*tile0=*/32 * NSEQ, /*tile1=*/1, /*stride0=*/32ull * NSEQ);
    __builtin_amdgcn_s_wait_tensorcnt(0);
  }
#else
  // ---- fallback: cooperative VGPR staging
  for (int it = tid; it < 1024; it += 64) {
    int k = it >> 2, c = it & 3;
    *(v8h*)&Ksh[k][c * 8] =
        *((const v8h*)(Kw + (size_t)(i * NSEQ + k) * 128 + h * 32) + c);
  }
  {
    const v8h* vs = (const v8h*)(VTw + (size_t)(i * 4 + h) * 32 * NSEQ);
    v8h* vd = (v8h*)&VTsh[0][0];
    for (int it = tid; it < 1024; it += 64) vd[it] = vs[it];
  }
#endif
  __syncthreads();

#pragma unroll 1
  for (int t = 0; t < 8; ++t) {           // each wave: 8 j-tiles
    int jt = wave + 2 * t;
    int jbase = jt * 16;

    // Q fragment for this j-tile (Q pre-scaled by 1/sqrt(d))
    const _Float16* qrow = Qw + (size_t)(i * NSEQ + jbase) * 128 + h * 32;
    v16h aq = load_A16(qrow, 128, 0, lane);
    if (t < 7) {  // prefetch next j-tile's Q rows
      __builtin_prefetch(qrow + 2 * 16 * 128, 0, 0);
    }

    // scores = bias + Q K^T  (acc initialized from bias in D-layout)
    v8f sc[16];
#pragma unroll
    for (int kt = 0; kt < 16; ++kt) {
      v8f acc;
#pragma unroll
      for (int v = 0; v < 8; ++v) {
        int jr = jbase + v + 8 * hi;
        acc[v] = biasw[(size_t)h * NPOS + (size_t)jr * NSEQ + kt * 16 + n];
      }
      int kcol = kt * 16 + n;
      v16h bk = load16(&Ksh[kcol][hi * 16]);
      sc[kt] = wmma16(aq, bk, acc);
    }

    // row-wise softmax over 256 k (16 in-lane tiles x 16 lanes)
#pragma unroll
    for (int v = 0; v < 8; ++v) {
      float m = -3.0e38f;
#pragma unroll
      for (int kt = 0; kt < 16; ++kt) m = fmaxf(m, sc[kt][v]);
      m = red16max(m);
      float ssum = 0.f;
#pragma unroll
      for (int kt = 0; kt < 16; ++kt) {
        float e = __expf(sc[kt][v] - m);
        sc[kt][v] = e;
        ssum += e;
      }
      ssum = red16sum(ssum);
      float inv = 1.f / ssum;
#pragma unroll
      for (int kt = 0; kt < 16; ++kt) sc[kt][v] *= inv;
    }

    // P -> LDS (f16) for the D-layout -> A-layout transpose
#pragma unroll
    for (int kt = 0; kt < 16; ++kt) {
#pragma unroll
      for (int v = 0; v < 8; ++v) {
        Psh[wave][v + 8 * hi][kt * 16 + n] = (_Float16)sc[kt][v];
      }
    }

    // vals = P @ V  (K=256 => 8 WMMA steps; d=32 => 2 output tiles)
#pragma unroll
    for (int dt = 0; dt < 2; ++dt) {
      v8f acc = {};
#pragma unroll
      for (int k2 = 0; k2 < 8; ++k2) {
        v16h ap = load_A16(&Psh[wave][0][0], NSEQ, k2 * 32, lane);
        v16h bv = load16(&VTsh[dt * 16 + n][k2 * 32 + hi * 16]);
        acc = wmma16(ap, bv, acc);
      }
#pragma unroll
      for (int v = 0; v < 8; ++v) {
        int jr = jbase + v + 8 * hi;
        size_t pos = (size_t)(i * NSEQ + jr) * 128 + h * 32 + dt * 16 + n;
        float g = (float)Gw[pos];
        valsw[pos] = (_Float16)(acc[v] * g);
      }
    }
  }
}

// -------------------- 4) output projection: vals @ Wo + bo -----------------
__global__ void __launch_bounds__(128)
outproj_kernel(const _Float16* __restrict__ valsw,
               const _Float16* __restrict__ packed,
               const float* __restrict__ bo, float* __restrict__ out) {
  int wave = threadIdx.x >> 5, lane = threadIdx.x & 31;
  int tile = blockIdx.x * 4 + wave;
  int rowbase = tile * 16;
  int n = lane & 15, hi = lane >> 4;
  const _Float16* arow = valsw + (size_t)rowbase * 128;
  const _Float16* pw = packed + PK_O;
#pragma unroll
  for (int nt = 0; nt < 8; ++nt) {
    float bov = bo[nt * 16 + n];
    v8f acc;
#pragma unroll
    for (int v = 0; v < 8; ++v) acc[v] = bov;
#pragma unroll
    for (int kt = 0; kt < 4; ++kt) {
      v16h a = load_A16(arow, 128, kt * 32, lane);
      v16h b = load16(pw + ((size_t)(nt * 4 + kt) * 32 + lane) * 16);
      acc = wmma16(a, b, acc);
    }
#pragma unroll
    for (int v = 0; v < 8; ++v) {
      out[(size_t)(rowbase + v + 8 * hi) * 128 + nt * 16 + n] = acc[v];
    }
  }
}

// -------------------- launch ------------------------------------------------
extern "C" void kernel_launch(void* const* d_in, const int* in_sizes, int n_in,
                              void* d_out, int out_size, void* d_ws,
                              size_t ws_size, hipStream_t stream) {
  (void)in_sizes; (void)n_in; (void)out_size; (void)ws_size;
  const float* x     = (const float*)d_in[0];
  const float* gamma = (const float*)d_in[1];
  const float* beta  = (const float*)d_in[2];
  const float* Wq    = (const float*)d_in[3];
  const float* Wk    = (const float*)d_in[4];
  const float* Wv    = (const float*)d_in[5];
  const float* Wb    = (const float*)d_in[6];
  const float* Wg    = (const float*)d_in[7];
  const float* bg    = (const float*)d_in[8];
  const float* Wo    = (const float*)d_in[9];
  const float* bo    = (const float*)d_in[10];
  float* out = (float*)d_out;

  char* ws = (char*)d_ws;
  const size_t MB16 = (size_t)1 << 24;
  _Float16* packed = (_Float16*)ws;                          // ~164 KB
  char* buf = ws + ((size_t)1 << 18);
  _Float16* Qw    = (_Float16*)(buf + 0 * MB16);
  _Float16* Kw    = (_Float16*)(buf + 1 * MB16);
  _Float16* VTw   = (_Float16*)(buf + 2 * MB16);
  _Float16* Gw    = (_Float16*)(buf + 3 * MB16);
  _Float16* valsw = (_Float16*)(buf + 4 * MB16);
  float*    biasw = (float*)   (buf + 5 * MB16);             // 1 MB

  pack_kernel<<<384, 256, 0, stream>>>(Wq, Wk, Wv, Wg, Wo, Wb, packed);
  lnproj_kernel<<<1024, 128, 0, stream>>>(x, gamma, beta, packed, bg,
                                          Qw, Kw, VTw, Gw, biasw);
  attn_kernel<<<1024, 64, 0, stream>>>(Qw, Kw, VTw, Gw, biasw, valsw);
  outproj_kernel<<<1024, 128, 0, stream>>>(valsw, packed, bo, out);
}
